// LieNet_12283606467802
// MI455X (gfx1250) — matrix operand order repaired and checked by reading, not compiled, not run on hardware
//
#include <hip/hip_runtime.h>
#include <hip/hip_bf16.h>
#include <math.h>

// ---------------------------------------------------------------------------
// LieConv net for MI455X (gfx1250, wave32).
// d_in layout (JAX pytree flattening, sorted dict keys inside params):
//   0: pos [64,1024,2] f32   1: x (dead)   2: bn0_b  3: bn0_g (dead)
//   conv1 @4, conv2 @28, conv3 @52, 24 leaves each:
//     global[0](W,b,g,be) global[1] local[0] local[1] weight[0] weight[1]
//   76: lin_W 77: lin_b 78: mlp_W 79: mlp_b
// Output: [64,2] f32 softmax.
//
// All matmuls: v_wmma_f32_16x16x32_f16, guard-free b128 operand loads.
// A-operands stored with zero-padded K-stride (multiple of 32 halves);
// weights pre-transposed to f16 WT[Npad][Kpad]; einsum staged via LDS.
// ---------------------------------------------------------------------------

typedef __attribute__((ext_vector_type(16))) _Float16 v16h;
typedef __attribute__((ext_vector_type(8)))  float    v8f;
typedef __attribute__((ext_vector_type(8)))  _Float16 h8v;

#define NB 64
#define FLT_BIG 3.402823466e38f

// ---------------- farthest point sampling: one block per graph -------------
__global__ void fps_kernel(const float* __restrict__ pos, int n, int m,
                           int* __restrict__ idx) {
  int b = blockIdx.x;
  const float* p = pos + (size_t)b * n * 2;
  __shared__ float px[1024], py[1024], mind[1024];
  __shared__ float red_v[256];
  __shared__ int   red_i[256];
  __shared__ int   s_last;
  for (int j = threadIdx.x; j < n; j += blockDim.x) {
    px[j] = p[2 * j]; py[j] = p[2 * j + 1]; mind[j] = FLT_BIG;
  }
  if (threadIdx.x == 0) { idx[(size_t)b * m] = 0; s_last = 0; }
  __syncthreads();
  for (int s = 1; s < m; ++s) {
    int last = s_last;
    float lx = px[last], ly = py[last];
    float bestv = -1.0f; int besti = 0;
    for (int j = threadIdx.x; j < n; j += blockDim.x) {
      float dx = px[j] - lx, dy = py[j] - ly;
      float d = dx * dx + dy * dy;
      float md = mind[j]; md = md < d ? md : d; mind[j] = md;
      if (md > bestv) { bestv = md; besti = j; }   // strict > keeps lowest j
    }
    red_v[threadIdx.x] = bestv; red_i[threadIdx.x] = besti;
    __syncthreads();
    for (int off = blockDim.x >> 1; off > 0; off >>= 1) {
      if ((int)threadIdx.x < off) {
        float v2 = red_v[threadIdx.x + off]; int i2 = red_i[threadIdx.x + off];
        if (v2 > red_v[threadIdx.x] ||
            (v2 == red_v[threadIdx.x] && i2 < red_i[threadIdx.x])) {
          red_v[threadIdx.x] = v2; red_i[threadIdx.x] = i2;
        }
      }
      __syncthreads();
    }
    if (threadIdx.x == 0) { s_last = red_i[0]; idx[(size_t)b * m + s] = red_i[0]; }
    __syncthreads();
  }
}

// ---------------- gather sampled positions ---------------------------------
__global__ void gather_pos_kernel(const float* __restrict__ pin,
                                  const int* __restrict__ idx,
                                  int n, int m, float* __restrict__ pout,
                                  int total) {
  int t = blockIdx.x * blockDim.x + threadIdx.x;
  if (t >= total) return;
  int b = t / m;
  int j = idx[t];
  pout[2 * t]     = pin[((size_t)b * n + j) * 2];
  pout[2 * t + 1] = pin[((size_t)b * n + j) * 2 + 1];
}

// ---------------- kNN: one wave per center, LDS distances ------------------
__global__ void knn_kernel(const float* __restrict__ posc,
                           const float* __restrict__ pos,
                           int m, int n, int k, int* __restrict__ nbr,
                           int totalBM) {
  __shared__ float sd[8 * 1024];
  int wave = threadIdx.x >> 5;
  int lane = threadIdx.x & 31;
  int wg = blockIdx.x * 8 + wave;
  if (wg >= totalBM) return;                     // wave-uniform exit
  int b = wg / m, i = wg % m;
  float cx = posc[((size_t)b * m + i) * 2];
  float cy = posc[((size_t)b * m + i) * 2 + 1];
  float* d = sd + wave * 1024;
  for (int j = lane; j < n; j += 32) {
    float dx = pos[((size_t)b * n + j) * 2]     - cx;
    float dy = pos[((size_t)b * n + j) * 2 + 1] - cy;
    d[j] = dx * dx + dy * dy;
  }
  for (int kk = 0; kk < k; ++kk) {
    float bv = FLT_BIG; int bj = 0;
    for (int j = lane; j < n; j += 32) {
      float v = d[j];
      if (v < bv) { bv = v; bj = j; }
    }
    for (int off = 16; off > 0; off >>= 1) {
      float ov = __shfl_xor(bv, off, 32);
      int   oj = __shfl_xor(bj, off, 32);
      if (ov < bv || (ov == bv && oj < bj)) { bv = ov; bj = oj; }
    }
    if (lane == (bj & 31)) d[bj] = FLT_BIG;      // owning lane, no barrier
    if (lane == 0) nbr[(size_t)wg * k + kk] = bj;
  }
}

// ---------------- pair embedding + feature gather (zero-padded f16) --------
__global__ void edge_prep_kernel(const float* __restrict__ posc,
                                 const float* __restrict__ posn,
                                 const int* __restrict__ nbr,
                                 const _Float16* __restrict__ xprev,
                                 int Cin, int m, int n, int k, int F, int Fpad,
                                 int isFirst,
                                 _Float16* __restrict__ rel,   // stride 32
                                 _Float16* __restrict__ feat,  // stride Fpad
                                 long long total) {
  long long t = blockIdx.x * (long long)blockDim.x + threadIdx.x;
  if (t >= total) return;
  long long bi = t / k;
  int i = (int)(bi % m), b = (int)(bi / m);
  int j = nbr[t];
  float cx = posc[((size_t)b * m + i) * 2],  cy = posc[((size_t)b * m + i) * 2 + 1];
  float jx = posn[((size_t)b * n + j) * 2],  jy = posn[((size_t)b * n + j) * 2 + 1];
  float thc = atan2f(cy, cx), thj = atan2f(jy, jx);
  float rc = sqrtf(cx * cx + cy * cy), rj = sqrtf(jx * jx + jy * jy);
  float r0 = thj - thc;
  _Float16* rr = rel + (size_t)t * 32;
  rr[0] = (_Float16)r0; rr[1] = (_Float16)rc; rr[2] = (_Float16)rj;
  for (int c = 3; c < 32; ++c) rr[c] = (_Float16)0.0f;
  _Float16* f = feat + (size_t)t * Fpad;
  if (isFirst) {
    f[0] = (_Float16)jx; f[1] = (_Float16)jy;
    f[2] = (_Float16)r0; f[3] = (_Float16)rc; f[4] = (_Float16)rj;
  } else {
    const _Float16* xj = xprev + ((size_t)b * n + j) * Cin;
    for (int c = 0; c < Cin; ++c) f[c] = xj[c];
    f[Cin + 0] = (_Float16)r0; f[Cin + 1] = (_Float16)rc; f[Cin + 2] = (_Float16)rj;
  }
  for (int c = F; c < Fpad; ++c) f[c] = (_Float16)0.0f;
}

// ---------------- weight prep: W[K,N] f32 -> WT[NpadWT][Kpad] f16 ----------
__global__ void wprep_kernel(const float* __restrict__ W,
                             _Float16* __restrict__ WT,
                             int K, int N, int Kpad, int NpadWT) {
  int t = blockIdx.x * blockDim.x + threadIdx.x;
  if (t >= NpadWT * Kpad) return;
  int col = t / Kpad, kk = t % Kpad;
  float v = (col < N && kk < K) ? W[(size_t)kk * N + col] : 0.0f;
  WT[t] = (_Float16)v;
}

// ---------------- WMMA GEMM: A[M,Kpad] x WT[.,Kpad] -> C[M,Nstride] --------
// Guard-free operand loads: M%16==0, Kpad%32==0, zero-padded operands.
// (bias dropped: training-mode BN centering cancels it exactly)
__global__ void gemm_wmma_f16(const _Float16* __restrict__ A,
                              const _Float16* __restrict__ WT,
                              _Float16* __restrict__ C,
                              int M, int Kpad, int Nstride) {
  const int lane = threadIdx.x & 31;
  const int wave = threadIdx.x >> 5;
  const int mtile = blockIdx.x * 8 + wave;
  if (mtile * 16 >= M) return;                   // wave-uniform
  const int half = lane >> 4;
  const int lm   = lane & 15;
  const int row  = mtile * 16 + lm;              // always < M
  const int col  = blockIdx.y * 16 + lm;
  const _Float16* arow = A  + (size_t)row * Kpad;
  const _Float16* bcol = WT + (size_t)col * Kpad;
  v8f acc = {};
  for (int k0 = 0; k0 < Kpad; k0 += 32) {
    h8v alo = *(const h8v*)(arow + k0 + half * 8);        // K = k0+half*8 ..+8
    h8v ahi = *(const h8v*)(arow + k0 + 16 + half * 8);   // K = k0+16+half*8
    h8v blo = *(const h8v*)(bcol + k0 + half * 8);        // K = k0+half*8 ..+8
    h8v bhi = *(const h8v*)(bcol + k0 + 16 + half * 8);
    v16h a, b;
    #pragma unroll
    for (int e = 0; e < 8; ++e) {
      a[e] = alo[e]; a[e + 8] = ahi[e];
      b[e] = blo[e]; b[e + 8] = bhi[e];
    }
    acc = __builtin_amdgcn_wmma_f32_16x16x32_f16(false, a, false, b,
                                                 (short)0, acc, false, false);
  }
  #pragma unroll
  for (int r = 0; r < 8; ++r) {
    int orow = mtile * 16 + r + half * 8;
    if (col < Nstride) C[(size_t)orow * Nstride + col] = (_Float16)acc[r];
  }
}

// ---------------- BN helpers ----------------------------------------------
__global__ void zero_f(float* p, int n) {
  int t = blockIdx.x * blockDim.x + threadIdx.x;
  if (t < n) p[t] = 0.0f;
}

__global__ void bn_stats(const _Float16* __restrict__ X, long long total,
                         int Cs, int C, float* __restrict__ stats) {
  __shared__ float s[128];
  for (int i = threadIdx.x; i < 2 * C; i += blockDim.x) s[i] = 0.0f;
  __syncthreads();
  long long stride = (long long)gridDim.x * blockDim.x;
  for (long long t = blockIdx.x * (long long)blockDim.x + threadIdx.x;
       t < total; t += stride) {
    int c = (int)(t % Cs);
    if (c < C) {
      float v = (float)X[t];
      atomicAdd(&s[c], v);
      atomicAdd(&s[C + c], v * v);
    }
  }
  __syncthreads();
  for (int i = threadIdx.x; i < 2 * C; i += blockDim.x) atomicAdd(&stats[i], s[i]);
}

__global__ void bn_norm_relu(_Float16* __restrict__ X, long long total, int Cs,
                             int C, const float* __restrict__ stats,
                             const float* __restrict__ g,
                             const float* __restrict__ be, float invRows) {
  long long t = blockIdx.x * (long long)blockDim.x + threadIdx.x;
  if (t >= total) return;
  int c = (int)(t % Cs);
  if (c >= C) return;                            // pad stays zero
  float mean = stats[c] * invRows;
  float var  = stats[C + c] * invRows - mean * mean;
  float x = (float)X[t];
  float y = (x - mean) * rsqrtf(var + 1e-5f) * g[c] + be[c];
  X[t] = (_Float16)fmaxf(y, 0.0f);
}

// ---------------- einsum bmkc,bmkd->bmcd (/k) via WMMA + LDS staging -------
// one wave per center: stage h tile [k][Chpad] and w tile [k][8] into LDS,
// then transposed-A fragments come from conflict-free DS reads.
__global__ void einsum_wmma(const _Float16* __restrict__ h,
                            const _Float16* __restrict__ w,
                            _Float16* __restrict__ agg, int k, int Ch,
                            int Chpad) {
  __shared__ __align__(16) _Float16 sh[4096];    // up to 64x64
  __shared__ __align__(16) _Float16 sw[512];     // up to 64x8
  const int bm = blockIdx.x;
  const int lane = threadIdx.x;                  // 32 threads = 1 wave
  {
    const h8v* src = (const h8v*)(h + (size_t)bm * k * Chpad);
    h8v* dst = (h8v*)sh;
    int tot = (k * Chpad) >> 3;
    for (int i = lane; i < tot; i += 32) dst[i] = src[i];
    const h8v* srcw = (const h8v*)(w + (size_t)bm * k * 8);
    h8v* dstw = (h8v*)sw;
    for (int i = lane; i < k; i += 32) dstw[i] = srcw[i];
  }
  __syncthreads();                               // single wave: cheap fence
  const int half = lane >> 4, lm = lane & 15;
  const int nct = (Ch + 15) / 16;
  const float invk = 1.0f / (float)k;
  for (int ct = 0; ct < nct; ++ct) {
    v8f acc = {};
    for (int k0 = 0; k0 < k; k0 += 32) {
      v16h a, b;
      #pragma unroll
      for (int e = 0; e < 16; ++e) {
        int kk = k0 + ((e < 8) ? (half * 8 + e) : (16 + half * 8 + (e - 8)));
        a[e] = sh[kk * Chpad + ct * 16 + lm];
      }
      #pragma unroll
      for (int e = 0; e < 16; ++e) {
        int kk = k0 + ((e < 8) ? (half * 8 + e) : (16 + half * 8 + (e - 8)));
        b[e] = sw[kk * 8 + (lm & 7)];            // cols>=4 are zero/garbage:
      }                                          // they never reach stored D
      acc = __builtin_amdgcn_wmma_f32_16x16x32_f16(false, a, false, b,
                                                   (short)0, acc, false, false);
    }
    #pragma unroll
    for (int r = 0; r < 8; ++r) {
      int cc = ct * 16 + r + half * 8;
      if (cc < Ch && lm < 4)
        agg[(size_t)bm * (Ch * 4) + cc * 4 + lm] = (_Float16)(acc[r] * invk);
    }
  }
}

// ---------------- head: mean-pool + FC+ReLU + FC + softmax -----------------
__global__ void head_kernel(const _Float16* __restrict__ act,
                            const float* __restrict__ mlpW,
                            const float* __restrict__ mlpb,
                            const float* __restrict__ linW,
                            const float* __restrict__ linb,
                            float* __restrict__ out, int m) {
  int b = blockIdx.x, c = threadIdx.x;           // 64 threads
  __shared__ float mean[64], h1[64], o[2];
  float s = 0.0f;
  for (int i = 0; i < m; ++i) s += (float)act[((size_t)b * m + i) * 64 + c];
  mean[c] = s / (float)m;
  __syncthreads();
  float a = mlpb[c];
  for (int j = 0; j < 64; ++j) a += mean[j] * mlpW[j * 64 + c];
  h1[c] = fmaxf(a, 0.0f);
  __syncthreads();
  if (c < 2) {
    float v = linb[c];
    for (int j = 0; j < 64; ++j) v += h1[j] * linW[j * 2 + c];
    o[c] = v;
  }
  __syncthreads();
  if (c == 0) {
    float mx = fmaxf(o[0], o[1]);
    float e0 = expf(o[0] - mx), e1 = expf(o[1] - mx);
    float inv = 1.0f / (e0 + e1);
    out[b * 2]     = e0 * inv;
    out[b * 2 + 1] = e1 * inv;
  }
}

// ---------------------------------------------------------------------------
extern "C" void kernel_launch(void* const* d_in, const int* in_sizes, int n_in,
                              void* d_out, int out_size, void* d_ws, size_t ws_size,
                              hipStream_t stream) {
  (void)in_sizes; (void)out_size;
  if (n_in < 80) return;
  const float* pos0 = (const float*)d_in[0];
  auto Pf = [&](int i) { return (const float*)d_in[i]; };
  auto ceil32 = [](int v) { return (v + 31) & ~31; };

  // --- deterministic bump allocator over d_ws (peak ~615 MB) ---
  char* base = (char*)d_ws;
  size_t off = 0;
  auto alloc = [&](size_t bytes) -> void* {
    void* p = base + off;
    off = (off + bytes + 255) & ~(size_t)255;
    return p;
  };
  const size_t rowsE1 = 64u * 768 * 32;   // 1,572,864 edges L1 (max rows)
  const size_t rowsE3 = 64u * 346 * 64;   // 1,417,216 edges L3
  // feat (stride Fpad<=96), also reused for h (stride <=64)
  _Float16* featb = (_Float16*)alloc(rowsE3 * 96 * 2);
  // R1: rel (stride 32) early, local-hidden bufB (stride <=64) later
  _Float16* R1    = (_Float16*)alloc(rowsE3 * 64 * 2);
  _Float16* bufW1 = (_Float16*)alloc(rowsE1 * 32 * 2);  // weight hidden
  _Float16* bufW2 = (_Float16*)alloc(rowsE1 * 8 * 2);   // w, stride 8
  int*      nbr   = (int*)alloc(rowsE1 * 4);
  _Float16* agg   = (_Float16*)alloc((size_t)64 * 346 * 256 * 2);
  _Float16* bufG1 = (_Float16*)alloc((size_t)64 * 576 * 64 * 2);
  _Float16* x1    = (_Float16*)alloc((size_t)64 * 768 * 32 * 2);
  _Float16* x2    = (_Float16*)alloc((size_t)64 * 576 * 64 * 2);
  _Float16* x3    = (_Float16*)alloc((size_t)64 * 346 * 64 * 2);
  float* pos1 = (float*)alloc((size_t)64 * 768 * 2 * 4);
  float* pos2 = (float*)alloc((size_t)64 * 576 * 2 * 4);
  float* pos3 = (float*)alloc((size_t)64 * 346 * 2 * 4);
  int* idx1 = (int*)alloc((size_t)64 * 768 * 4);
  int* idx2 = (int*)alloc((size_t)64 * 576 * 4);
  int* idx3 = (int*)alloc((size_t)64 * 346 * 4);
  _Float16* wtArena = (_Float16*)alloc(262144);         // transposed weights
  float* stats = (float*)alloc(512);
  if (off > ws_size) return;   // workspace too small: cannot run
  size_t wtOff = 0;            // reset every call: deterministic

  // linear (K zero-padded) + train-mode BN + ReLU, in place on `out`
  auto runLin = [&](const _Float16* A, int M, int Kpad, int pW, int K, int N,
                    int Nstride, _Float16* out) {
    const float* W  = Pf(pW);
    const float* g  = Pf(pW + 2);
    const float* be = Pf(pW + 3);
    int NpadWT = Nstride < 16 ? 16 : Nstride;
    _Float16* WT = wtArena + wtOff;
    wtOff += (size_t)NpadWT * Kpad;
    int wtot = NpadWT * Kpad;
    wprep_kernel<<<(wtot + 255) / 256, 256, 0, stream>>>(W, WT, K, N, Kpad,
                                                         NpadWT);
    zero_f<<<1, 128, 0, stream>>>(stats, 2 * N);
    dim3 grid((M / 16 + 7) / 8, (Nstride + 15) / 16);
    gemm_wmma_f16<<<grid, 256, 0, stream>>>(A, WT, out, M, Kpad, Nstride);
    long long total = (long long)M * Nstride;
    bn_stats<<<1024, 256, 0, stream>>>(out, total, Nstride, N, stats);
    bn_norm_relu<<<(int)((total + 255) / 256), 256, 0, stream>>>(
        out, total, Nstride, N, stats, g, be, 1.0f / (float)M);
  };

  auto runConv = [&](const float* posPrev, float* posCur, int* idx,
                     int n, int m, int k, const _Float16* xPrev, int Cin,
                     int F, int Fpad, int H, int Cout, int Gh, int Gout,
                     int pbase, _Float16* xCur) {
    int HS = ceil32(H), CS = ceil32(Cout), GS = ceil32(Gh), OS = ceil32(Gout);
    fps_kernel<<<NB, 256, 0, stream>>>(posPrev, n, m, idx);
    int bm = NB * m;
    gather_pos_kernel<<<(bm + 255) / 256, 256, 0, stream>>>(posPrev, idx, n, m,
                                                            posCur, bm);
    knn_kernel<<<(bm + 7) / 8, 256, 0, stream>>>(posCur, posPrev, m, n, k, nbr,
                                                 bm);
    long long rows = (long long)bm * k;
    edge_prep_kernel<<<(int)((rows + 255) / 256), 256, 0, stream>>>(
        posCur, posPrev, nbr, xPrev, Cin, m, n, k, F, Fpad,
        (Cin == 0) ? 1 : 0, R1 /*rel*/, featb, rows);
    // weight MLP: 3 -> 8 -> 4        (rel lives in R1, dead after lin1)
    runLin(R1,    (int)rows, 32, pbase + 16, 3, 8, 32, bufW1);
    runLin(bufW1, (int)rows, 32, pbase + 20, 8, 4, 8,  bufW2);
    // local MLP: F -> H -> Cout      (bufB reuses R1; h reuses featb)
    runLin(featb, (int)rows, Fpad, pbase + 8,  F, H,    HS, R1);
    runLin(R1,    (int)rows, HS,   pbase + 12, H, Cout, CS, featb);
    // einsum bmkc,bmkd->bmcd / k  -> agg [bm, Cout*4]
    einsum_wmma<<<bm, 32, 0, stream>>>(featb, bufW2, agg, k, Cout, CS);
    // global MLP: Cout*4 -> Gh -> Gout
    runLin(agg,   bm, Cout * 4, pbase + 0, Cout * 4, Gh,   GS, bufG1);
    runLin(bufG1, bm, GS,       pbase + 4, Gh,       Gout, OS, xCur);
  };

  // conv1 (is_first): n=1024 m=768 k=32, local 5->12->8, global 32->32->32
  runConv(pos0, pos1, idx1, 1024, 768, 32, nullptr, 0, 5, 32, 12, 8, 32, 32,
          4, x1);
  // conv2: n=768 m=576 k=32, local 35->32->32, global 128->64->64
  runConv(pos1, pos2, idx2, 768, 576, 32, x1, 32, 35, 64, 32, 32, 64, 64,
          28, x2);
  // conv3: n=576 m=346 k=64, local 67->64->64, global 256->64->64
  runConv(pos2, pos3, idx3, 576, 346, 64, x2, 64, 67, 96, 64, 64, 64, 64,
          52, x3);

  // head: mean over m3, mlp(64->64)+ReLU, lin(64->2), softmax
  head_kernel<<<NB, 64, 0, stream>>>(x3, Pf(78), Pf(79), Pf(76), Pf(77),
                                     (float*)d_out, 346);
}
